// OptimizedBandPassFilter_39333310496999
// MI455X (gfx1250) — matrix operand order repaired and checked by reading, not compiled, not run on hardware
//
#include <hip/hip_runtime.h>

typedef __attribute__((ext_vector_type(16))) __bf16 v16bf;
typedef __attribute__((ext_vector_type(8)))  float  v8f;
typedef __attribute__((ext_vector_type(4)))  unsigned int v4u;

#define SEQ      8192
#define NF       20
#define KLEN     769
#define KPAD     800                 // 25 chunks of K=32
#define NCHUNK   25
#define JPAD     (KPAD - KLEN)       // 31 leading zeros in reversed kernel
#define TBLK     512                 // time positions per block
#define TT_PER_W 4                   // 16-wide time tiles per wave
#define THREADS  256                 // 8 wave32
#define NFT      2                   // two 16-wide filter tiles (20 -> 32 padded)
#define XS_LEN   1312                // TBLK + KPAD - 1 = 1311, rounded even
#define XS_STRIDE 1328               // copy stride (u16): 16B-aligned, bank-spread
#define XSW      (XS_STRIDE / 2)     // copy stride in u32 units (664)

#define B_TILE_U16   (NCHUNK * 32 * 16)            // 12800 u16 per (ftile,hi/lo)
#define B_AREA_U16   (NFT * 2 * B_TILE_U16)        // 51200 u16 = 102400 B
#define X_AREA_U16   (16 * XS_STRIDE)              // 8 hi + 8 lo shifted copies
#define SMEM_BYTES   ((B_AREA_U16 + X_AREA_U16) * 2)   // 144896 B

__device__ __forceinline__ unsigned short bf16_rne(float f) {
  unsigned int u = __float_as_uint(f);
  u += 0x7FFFu + ((u >> 16) & 1u);   // round-to-nearest-even
  return (unsigned short)(u >> 16);
}
__device__ __forceinline__ void split_bf16(float f, unsigned short& hi, unsigned short& lo) {
  hi = bf16_rne(f);
  lo = bf16_rne(f - __uint_as_float((unsigned int)hi << 16));
}

union AFrag { v4u x[2]; v16bf v; };

__global__ __launch_bounds__(THREADS) void conv1d_wmma_bf16x3(
    const float* __restrict__ x, const float* __restrict__ kern,
    float* __restrict__ out)
{
  // Dynamic LDS (144.9 KB; 2 blocks per 320 KB WGP):
  //   B images, interleaved per chunk: [c][ft*2+hl][lane][16 bf16]  (102400 B)
  //   x window: 8 shifted copies of hi then of lo, stride XS_STRIDE ( 42496 B)
  extern __shared__ __attribute__((aligned(32))) unsigned short smem[];
  unsigned short* sB = smem;
  unsigned short* sX = smem + B_AREA_U16;

  const int tid = threadIdx.x;
  const int b   = blockIdx.y;
  const int t0  = blockIdx.x * TBLK;

  // ---- Stage kernels into WMMA-B layout (Krev[f,j] = kern[f, KPAD-1-j]) ----
  for (int idx = tid; idx < NFT * B_TILE_U16; idx += THREADS) {
    const int ft = (idx >= B_TILE_U16);
    const int r  = idx - ft * B_TILE_U16;
    const int e = r & 15, L = (r >> 4) & 31, c = r >> 9;
    const int j = 32 * c + ((L & 16) ? 8 : 0) + (e & 7) + ((e & 8) ? 16 : 0);  // K index within padded 800
    const int f = ft * 16 + (L & 15);                // B column = filter
    float v = 0.f;
    if (f < NF && j >= JPAD) v = kern[f * KLEN + (KPAD - 1 - j)];
    unsigned short hi, lo; split_bf16(v, hi, lo);
    const int dst = ((c * 4 + ft * 2) * 32 + L) * 16 + e;   // img = ft*2 (hi)
    sB[dst]       = hi;
    sB[dst + 512] = lo;                                     // img = ft*2+1 (lo)
  }
  // ---- Stage x window: 8 shifted copies so every lane's A-quad is 16B-aligned
  for (int i = tid; i < XS_LEN; i += THREADS) {
    const int g = t0 - (KPAD - 1) + i;
    float v = (g >= 0 && g < SEQ && i < TBLK + KPAD - 1) ? x[(size_t)b * SEQ + g] : 0.f;
    unsigned short hi, lo; split_bf16(v, hi, lo);
#pragma unroll
    for (int s = 0; s < 8; ++s) {
      const int p = i - s;                 // copy s holds xs[p + s] at position p
      if (p >= 0) {
        sX[s * XS_STRIDE + p]                 = hi;
        sX[(8 + s) * XS_STRIDE + p]           = lo;
      }
    }
  }
  __syncthreads();

  const int lane  = tid & 31;
  const int w     = tid >> 5;
  const int m     = lane & 15;                 // A row (time) / D column (filter)
  const int khalf = (lane & 16) ? 8 : 0;       // A K-offset for upper half-wave

  // Lane base: copy s=m&7; u32 offset ((m&8)+khalf)/2 is a multiple of 4.
  const v4u* pA = (const v4u*)((const unsigned int*)sX
                               + (m & 7) * XSW + (((m & 8) + khalf) >> 1))
                  + w * (TT_PER_W * 2);
  const unsigned short* pB = sB + (lane << 4);

  v8f acc[TT_PER_W][NFT] = {};

  for (int c = 0; c < NCHUNK; ++c) {
    const v16bf B0h = *(const v16bf*)(pB + c * 2048);
    const v16bf B0l = *(const v16bf*)(pB + c * 2048 + 512);
    const v16bf B1h = *(const v16bf*)(pB + c * 2048 + 1024);
    const v16bf B1l = *(const v16bf*)(pB + c * 2048 + 1536);
#pragma unroll
    for (int q = 0; q < TT_PER_W; ++q) {
      // Hankel A tile: elements [i0, i0+7] and [i0+16, i0+23], both 16B-aligned.
      const v4u* pq = pA + c * 4 + q * 2;
      AFrag Ah, Al;
      Ah.x[0] = pq[0];
      Ah.x[1] = pq[2];
      Al.x[0] = pq[2 * XSW];                  // lo copies: +8*XS_STRIDE u16
      Al.x[1] = pq[2 * XSW + 2];
      // bf16x3 compensated products for both filter tiles (6 WMMA / A-build)
      acc[q][0] = __builtin_amdgcn_wmma_f32_16x16x32_bf16(false, Ah.v, false, B0h,
                                                          (short)0, acc[q][0], false, false);
      acc[q][0] = __builtin_amdgcn_wmma_f32_16x16x32_bf16(false, Ah.v, false, B0l,
                                                          (short)0, acc[q][0], false, false);
      acc[q][0] = __builtin_amdgcn_wmma_f32_16x16x32_bf16(false, Al.v, false, B0h,
                                                          (short)0, acc[q][0], false, false);
      acc[q][1] = __builtin_amdgcn_wmma_f32_16x16x32_bf16(false, Ah.v, false, B1h,
                                                          (short)0, acc[q][1], false, false);
      acc[q][1] = __builtin_amdgcn_wmma_f32_16x16x32_bf16(false, Ah.v, false, B1l,
                                                          (short)0, acc[q][1], false, false);
      acc[q][1] = __builtin_amdgcn_wmma_f32_16x16x32_bf16(false, Al.v, false, B1h,
                                                          (short)0, acc[q][1], false, false);
    }
  }

  // ---- Store: D element (M = r + 8*(lane>=16), N = lane%16) -> out[b,0,f,t] ----
  const int roff = (lane & 16) ? 8 : 0;
#pragma unroll
  for (int ft = 0; ft < NFT; ++ft) {
    const int f = ft * 16 + m;
    if (f < NF) {
#pragma unroll
      for (int q = 0; q < TT_PER_W; ++q) {
        const int t = t0 + (w * TT_PER_W + q) * 16 + roff;
        float* op = out + ((size_t)b * NF + f) * SEQ + t;
        *(float4*)(op)     = make_float4(acc[q][ft][0], acc[q][ft][1],
                                         acc[q][ft][2], acc[q][ft][3]);
        *(float4*)(op + 4) = make_float4(acc[q][ft][4], acc[q][ft][5],
                                         acc[q][ft][6], acc[q][ft][7]);
      }
    }
  }
}

extern "C" void kernel_launch(void* const* d_in, const int* in_sizes, int n_in,
                              void* d_out, int out_size, void* d_ws, size_t ws_size,
                              hipStream_t stream) {
  const float* x = (const float*)d_in[0];      // (B,1,8192) f32
  const float* k = (const float*)d_in[1];      // (20,769)  f32
  float* out = (float*)d_out;                  // (B,1,20,8192) f32
  const int B = in_sizes[0] / SEQ;

  // Allow >64 KB dynamic LDS (host-side attribute; not a stream op, so it is
  // safe and idempotent under hipGraph capture).
  (void)hipFuncSetAttribute(reinterpret_cast<const void*>(conv1d_wmma_bf16x3),
                            hipFuncAttributeMaxDynamicSharedMemorySize, SMEM_BYTES);

  dim3 grid(SEQ / TBLK, B, 1);                 // 16 x 128 blocks, 8 waves each
  conv1d_wmma_bf16x3<<<grid, dim3(THREADS), SMEM_BYTES, stream>>>(x, k, out);
}
